// MatchingLoss_19095424598034
// MI455X (gfx1250) — matrix operand order repaired and checked by reading, not compiled, not run on hardware
//
#include <hip/hip_runtime.h>
#include <hip/hip_fp16.h>

typedef __attribute__((ext_vector_type(16))) _Float16 v16h;
typedef __attribute__((ext_vector_type(8)))  float    v8f;
typedef unsigned int __attribute__((ext_vector_type(4))) u32x4;
typedef int          __attribute__((ext_vector_type(4))) i32x4;
typedef int          __attribute__((ext_vector_type(8))) i32x8;

#define D_    64
#define L_    64
#define S_    10
#define Q_    30
#define SPC_  40            // S_+Q_
#define DIM_  128
#define NSUP  (L_*S_)       // 640
#define NQRY  (L_*Q_)       // 1920
#define TOTAL (D_*L_*SPC_)  // 163840
#define QCHUNK 64           // queries per workgroup (4 waves x 16)
#define NCHUNK (NQRY/QCHUNK) // 30
#define NSLICE 20           // K-slices of 32 supports
#define SLAB_W 136          // 128 halves + 16B pad -> B-frag ds_load conflict-free
#define EBUF_W 40           // 32 halves + pad

#if defined(__AMDGCN__) && __has_builtin(__builtin_amdgcn_tensor_load_to_lds)
#define HAVE_TDM 1
#else
#define HAVE_TDM 0
#endif

union FragU { v16h v; uint4 u[2]; _Float16 h[16]; };

#if HAVE_TDM
// Issue one TDM descriptor: 32 support rows x 256B -> LDS, +16B pad per row.
__device__ __forceinline__ void tdm_issue(const _Float16* gsrc, uint32_t lds_off) {
    const uint64_t ga = (uint64_t)(size_t)gsrc;
    u32x4 g0 = { 1u,                                   // count=1, normal mode
                 lds_off,
                 (uint32_t)ga,
                 (uint32_t)((ga >> 32) & 0x01FFFFFFu) | 0x80000000u }; // type=2
    i32x8 g1 = { (int)((1u << 16)                      // data_size = 2 bytes
                     | (1u << 20)                      // pad_enable
                     | (5u << 22)                      // pad_interval: 64 dwords
                     | (3u << 25)),                    // pad_amount:   4 dwords
                 (int)(4096u << 16),                   // tensor_dim0 = 4096
                 (int)(1u << 16),                      // tensor_dim1 = 1
                 (int)(4096u << 16),                   // tile_dim0   = 4096
                 1,                                    // tile_dim1   = 1
                 4096,                                 // tensor_dim0_stride
                 0, 0 };
    i32x4 gz = { 0, 0, 0, 0 };
#if __clang_major__ >= 23
    i32x8 gz8 = { 0, 0, 0, 0, 0, 0, 0, 0 };
    __builtin_amdgcn_tensor_load_to_lds(g0, g1, gz, gz, gz8, 0);
#else
    __builtin_amdgcn_tensor_load_to_lds(g0, g1, gz, gz, 0);
#endif
}
#endif

// ---------------------------------------------------------------------------
// Kernel 1: L2-normalize fp32 rows -> f16, scatter into episode layout
// ---------------------------------------------------------------------------
__global__ void k_normalize(const float* __restrict__ emb,
                            _Float16* __restrict__ sn,
                            _Float16* __restrict__ qn) {
    const int row  = blockIdx.x * 8 + (threadIdx.x >> 5);   // 1 wave per row
    const int lane = threadIdx.x & 31;
    const float* src = emb + (size_t)row * DIM_;

    float4 v = *(const float4*)(src + lane * 4);
    float ss = v.x*v.x + v.y*v.y + v.z*v.z + v.w*v.w;
    #pragma unroll
    for (int off = 16; off > 0; off >>= 1) ss += __shfl_xor(ss, off, 32);
    const float scale = 1.0f / (sqrtf(ss) + 1e-8f);

    const int d   = row / (L_ * SPC_);
    const int e   = row % (L_ * SPC_);
    const int cls = e / SPC_;
    const int k   = e % SPC_;
    _Float16* dst;
    if (k < S_) dst = sn + ((size_t)d * NSUP + cls * S_ + k)        * DIM_;
    else        dst = qn + ((size_t)d * NQRY + cls * Q_ + (k - S_)) * DIM_;

    union { _Float16 h[4]; uint2 u; } o;
    o.h[0] = (_Float16)(v.x * scale);
    o.h[1] = (_Float16)(v.y * scale);
    o.h[2] = (_Float16)(v.z * scale);
    o.h[3] = (_Float16)(v.w * scale);
    *(uint2*)(dst + lane * 4) = o.u;
}

// ---------------------------------------------------------------------------
// Kernel 2: per-episode WMMA GEMM1 (cosine sims) + WMMA GEMM2 (one-hot class
// reduction of exp(sim)) + NLL/argmax.  128 threads = 4 waves, 16 q-rows each.
// Support slabs double-buffered in LDS, fed by pipelined TDM transfers:
// issue slice jj+1, s_wait_tensorcnt(1) -> slice jj ready, DMA overlaps WMMA.
// ---------------------------------------------------------------------------
__global__ void k_episode(const _Float16* __restrict__ sn,
                          const _Float16* __restrict__ qn,
                          const int* __restrict__ labels,
                          float* __restrict__ partials) {
    __shared__ _Float16 slab[2][32][SLAB_W];     // ping-pong support slices
    __shared__ _Float16 Ebuf[4][16][EBUF_W];     // per-wave exp(sim) transpose buf
    __shared__ float    predAcc[4][16][L_];      // [wave][row][class]
    __shared__ float    wgSum[2];

    const int tid   = threadIdx.x;
    const int wave  = tid >> 5;
    const int lane  = tid & 31;
    const int d     = blockIdx.x / NCHUNK;
    const int chunk = blockIdx.x % NCHUNK;

    if (tid == 0) { wgSum[0] = 0.0f; wgSum[1] = 0.0f; }

    const int qbase = chunk * QCHUNK + wave * 16;
    const int m     = lane & 15;            // A frag M / B frag N index
    const int khalf = lane >> 4;
    const int koff8 = khalf * 8;            // A frag: K {koff8..+7, koff8+16..+23}
    const int bkoff = khalf * 16;           // unused; B frags use A's koff8 K split
    const int n     = m;

    // Hoist query A fragments for all 4 k-steps (K = 128 = 4 x 32)
    const _Float16* qrow = qn + ((size_t)d * NQRY + qbase + m) * DIM_;
    FragU a[4];
    #pragma unroll
    for (int t = 0; t < 4; ++t) {
        a[t].u[0] = *(const uint4*)(qrow + t * 32 + koff8);
        a[t].u[1] = *(const uint4*)(qrow + t * 32 + 16 + koff8);
    }

    const _Float16* sb = sn + (size_t)d * NSUP * DIM_;

#if HAVE_TDM
    if (wave == 0)                           // prime the pipeline: slice 0
        tdm_issue(sb, (uint32_t)(size_t)(&slab[0][0][0]));
#endif

    for (int g = 0; g < 4; ++g) {           // 4 class tiles of 16 classes
        v8f c2 = {};                        // pred accumulator [16q x 16cls]
        for (int jf = 0; jf < 5; ++jf) {    // 5 K-slices of 32 supports
            const int jj = g * 5 + jf;
            const int buf = jj & 1;

            __syncthreads();                // all reads of both buffers done
#if HAVE_TDM
            if (wave == 0) {
                if (jj < NSLICE - 1) {      // overlap next DMA with this compute
                    tdm_issue(sb + (size_t)(jj + 1) * 32 * DIM_,
                              (uint32_t)(size_t)(&slab[buf ^ 1][0][0]));
                    __builtin_amdgcn_s_wait_tensorcnt(1);  // slice jj complete
                } else {
                    __builtin_amdgcn_s_wait_tensorcnt(0);
                }
            }
#else
            {   // cooperative fallback: 128 threads x 64B into slab[buf]
                const int r  = tid >> 2;
                const int ch = tid & 3;
                const _Float16* gsrc = sb + (size_t)(jj * 32 + r) * DIM_ + ch * 32;
                uint4* dstp = (uint4*)&slab[buf][r][ch * 32];
                #pragma unroll
                for (int u = 0; u < 4; ++u) dstp[u] = *(const uint4*)(gsrc + u * 8);
            }
#endif
            if (tid == 0 && jj < NSLICE - 1)
                __builtin_prefetch(sb + (size_t)(jj + 1) * 32 * DIM_, 0, 1);
            __syncthreads();                // slab[buf] visible to all waves

            // ---- GEMM1: sims for 2 support tiles, exp -> Ebuf (f16) ----
            #pragma unroll
            for (int hv = 0; hv < 2; ++hv) {
                v8f c = {};
                #pragma unroll
                for (int t = 0; t < 4; ++t) {
                    FragU b;
                    b.u[0] = *(const uint4*)&slab[buf][hv * 16 + n][t * 32 + koff8];
                    b.u[1] = *(const uint4*)&slab[buf][hv * 16 + n][t * 32 + 16 + koff8];
                    c = __builtin_amdgcn_wmma_f32_16x16x32_f16(
                            false, a[t].v, false, b.v, (short)0, c, false, false);
                }
                // C layout: VGPR r -> row koff8+r, col n
                #pragma unroll
                for (int r = 0; r < 8; ++r)
                    Ebuf[wave][koff8 + r][hv * 16 + n] = (_Float16)__expf(c[r]);
            }

            // ---- GEMM2: A2 = exp tile (A layout), B2 = one-hot class mask ----
            FragU a2;
            a2.u[0] = *(const uint4*)&Ebuf[wave][m][koff8];
            a2.u[1] = *(const uint4*)&Ebuf[wave][m][16 + koff8];

            // class (g*16+n) owns supports [(g*16+n)*10, +10); local k = i<8 ? koff8+i : 16+koff8+(i-8)
            const int lo = (g * 16 + n) * 10 - jj * 32;
            FragU b2;
            #pragma unroll
            for (int i = 0; i < 16; ++i) {
                const int kk = (i < 8) ? (koff8 + i) : (16 + koff8 + i - 8);
                b2.h[i] = (kk >= lo && kk < lo + 10) ? (_Float16)1.0f : (_Float16)0.0f;
            }

            c2 = __builtin_amdgcn_wmma_f32_16x16x32_f16(
                     false, a2.v, false, b2.v, (short)0, c2, false, false);
        }
        // one writer per (row, class): plain stores, no atomics
        #pragma unroll
        for (int r = 0; r < 8; ++r)
            predAcc[wave][koff8 + r][g * 16 + n] = c2[r];
    }
    __syncthreads();

    // ---- epilogue: denom, argmax, NLL (lanes 0..15 per wave) ----
    if (lane < 16) {
        const int q = qbase + lane;
        float denom = 0.0f, best = -1.0f;
        int besti = 0;
        #pragma unroll 8
        for (int cc = 0; cc < L_; ++cc) {
            const float v = predAcc[wave][lane][cc];
            denom += v;
            if (v > best) { best = v; besti = cc; }
        }
        const int cls_q = q / Q_;
        const int qi    = q % Q_;
        const int lbl   = labels[(size_t)d * (L_ * SPC_) + cls_q * SPC_ + S_ + qi];
        float pv = predAcc[wave][lane][lbl] / denom;
        pv = fminf(fmaxf(pv, 1e-8f), 1.0f - 1e-8f);
        atomicAdd(&wgSum[0], -__logf(pv));
        atomicAdd(&wgSum[1], (besti == lbl) ? 1.0f : 0.0f);
    }
    __syncthreads();

    if (tid == 0) {
        partials[(size_t)blockIdx.x * 2 + 0] = wgSum[0];
        partials[(size_t)blockIdx.x * 2 + 1] = wgSum[1];
    }
}

// ---------------------------------------------------------------------------
// Kernel 3: reduce 1920 partials -> d_out = {loss, acc}
// ---------------------------------------------------------------------------
__global__ void k_reduce(const float* __restrict__ partials,
                         float* __restrict__ out) {
    __shared__ float s[2][256];
    const int tid = threadIdx.x;
    float nl = 0.0f, ac = 0.0f;
    for (int i = tid; i < D_ * NCHUNK; i += 256) {
        nl += partials[(size_t)i * 2 + 0];
        ac += partials[(size_t)i * 2 + 1];
    }
    s[0][tid] = nl; s[1][tid] = ac;
    __syncthreads();
    for (int w = 128; w > 0; w >>= 1) {
        if (tid < w) { s[0][tid] += s[0][tid + w]; s[1][tid] += s[1][tid + w]; }
        __syncthreads();
    }
    if (tid == 0) {
        const float inv = 1.0f / (float)(D_ * NQRY);
        out[0] = s[0][0] * inv;   // loss
        out[1] = s[1][0] * inv;   // acc
    }
}

// ---------------------------------------------------------------------------
extern "C" void kernel_launch(void* const* d_in, const int* in_sizes, int n_in,
                              void* d_out, int out_size, void* d_ws, size_t ws_size,
                              hipStream_t stream) {
    const float* emb    = (const float*)d_in[0];
    const int*   labels = (const int*)d_in[1];
    float* out = (float*)d_out;

    char* ws = (char*)d_ws;
    _Float16* sn = (_Float16*)ws;                                    // 10.5 MB
    _Float16* qn = (_Float16*)(ws + (size_t)D_ * NSUP * DIM_ * 2);   // 31.5 MB
    float* partials = (float*)(ws + (size_t)D_ * NSUP * DIM_ * 2
                                  + (size_t)D_ * NQRY * DIM_ * 2);   // 15 KB

    k_normalize<<<TOTAL / 8, 256, 0, stream>>>(emb, sn, qn);
    k_episode<<<D_ * NCHUNK, 128, 0, stream>>>(sn, qn, labels, partials);
    k_reduce<<<1, 256, 0, stream>>>(partials, out);
    (void)in_sizes; (void)n_in; (void)out_size; (void)ws_size;
}
